// DeepseekV3MoECalibrate_47725676593264
// MI455X (gfx1250) — matrix operand order repaired and checked by reading, not compiled, not run on hardware
//
#include <hip/hip_runtime.h>
#include <hip/hip_bf16.h>

// ---------------- problem constants (from reference) ----------------
#define T_TOK 2048   // B*S = 2*1024
#define H_DIM 2048
#define E_EXP 32
#define I_DIM 1024
#define IS_DIM 2048
#define TOPK 8
#define ROUTED_SCALING 2.5f

// ---------------- WMMA types ----------------
typedef float          __attribute__((ext_vector_type(8)))  v8f;
typedef __bf16         __attribute__((ext_vector_type(16))) v16bf;
typedef unsigned int   __attribute__((ext_vector_type(8)))  v8u;

__device__ __forceinline__ unsigned short f32_to_bf16(float f) {
    unsigned int u = __builtin_bit_cast(unsigned int, f);
    u += 0x7fffu + ((u >> 16) & 1u);   // round-to-nearest-even
    return (unsigned short)(u >> 16);
}

__device__ __forceinline__ unsigned int pack2_bf16(float lo, float hi) {
    return ((unsigned int)f32_to_bf16(hi) << 16) | (unsigned int)f32_to_bf16(lo);
}

// fast sigmoid via v_rcp_f32 (no IEEE div expansion)
__device__ __forceinline__ float fast_sigmoid(float x) {
    return __builtin_amdgcn_rcpf(1.f + __expf(-x));
}

// A fragment (16x32 bf16, MxK) per ISA layout, row-major global [M,K]:
// lanes 0-15 : M=lane,    elems 0..7 = K k0..k0+7,    elems 8..15 = K k0+16..k0+23
// lanes 16-31: M=lane-16, elems 0..7 = K k0+8..k0+15, elems 8..15 = K k0+24..k0+31
__device__ __forceinline__ v16bf load_a_frag(const unsigned short* __restrict__ A,
                                             int ld, int row0, int k0, int lane) {
    const unsigned short* p = A + (size_t)(row0 + (lane & 15)) * ld
                                + k0 + ((lane >> 4) << 3);
    uint4 lo = *(const uint4*)(p);
    uint4 hi = *(const uint4*)(p + 16);
    v8u t;
    t[0]=lo.x; t[1]=lo.y; t[2]=lo.z; t[3]=lo.w;
    t[4]=hi.x; t[5]=hi.y; t[6]=hi.z; t[7]=hi.w;
    return __builtin_bit_cast(v16bf, t);
}

// B fragment (32x16 bf16, KxN) from pre-transposed global Wt [N,K]:
// lanes 0-15: N=lane, K k0..k0+7, k0+16..k0+23; lanes 16-31: N=lane-16, K k0+8..k0+15, k0+24..k0+31
__device__ __forceinline__ v16bf load_b_frag(const unsigned short* __restrict__ Wt,
                                             int ld, int n0, int k0, int lane) {
    const unsigned short* p = Wt + (size_t)(n0 + (lane & 15)) * ld
                                 + k0 + ((lane >> 4) << 3);
    uint4 lo = *(const uint4*)(p);
    uint4 hi = *(const uint4*)(p + 16);
    v8u t;
    t[0]=lo.x; t[1]=lo.y; t[2]=lo.z; t[3]=lo.w;
    t[4]=hi.x; t[5]=hi.y; t[6]=hi.z; t[7]=hi.w;
    return __builtin_bit_cast(v16bf, t);
}

__device__ __forceinline__ v8f wmma_bf16(v16bf a, v16bf b, v8f c) {
    return __builtin_amdgcn_wmma_f32_16x16x32_bf16(
        false, a, false, b, (short)0, c, false, false);
}

// ---------------- kernel: fp32 -> bf16 row-major conversion (vectorized) ----------------
__global__ __launch_bounds__(256) void k_convert_bf16(const float4* __restrict__ x,
                                                      uint2* __restrict__ xb, int n4) {
    int i = blockIdx.x * 256 + threadIdx.x;
    if (i < n4) {
        float4 v = x[i];
        uint2 o;
        o.x = pack2_bf16(v.x, v.y);
        o.y = pack2_bf16(v.z, v.w);
        xb[i] = o;
    }
}

// ---------------- kernel: transpose + convert W[K,N] f32 -> Wt[N,K] bf16 ----------------
// 64x64 tiles via LDS; coalesced loads and packed 32-bit coalesced stores.
__global__ __launch_bounds__(256) void k_transpose_bf16(const float* __restrict__ W,
                                                        unsigned short* __restrict__ Wt,
                                                        int K, int N) {
    __shared__ float tile[64][65];
    const int n0 = blockIdx.x * 64;
    const int k0 = blockIdx.y * 64;
    const int tid = threadIdx.x;
    #pragma unroll
    for (int i = 0; i < 16; ++i) {
        int idx = tid + i * 256;        // 0..4095
        int kk = idx >> 6, nn = idx & 63;
        tile[kk][nn] = W[(size_t)(k0 + kk) * N + n0 + nn];
    }
    __syncthreads();
    #pragma unroll
    for (int i = 0; i < 8; ++i) {
        int idx = tid + i * 256;        // 0..2047
        int nn = idx >> 5;              // 0..63
        int kp = idx & 31;              // K-pair 0..31
        unsigned int packed = pack2_bf16(tile[2 * kp][nn], tile[2 * kp + 1][nn]);
        *(unsigned int*)(Wt + (size_t)(n0 + nn) * K + k0 + 2 * kp) = packed;
    }
}

// ---------------- kernel: router scores = sigmoid(x @ gateW^T) ----------------
__global__ __launch_bounds__(256) void k_router_scores(const float* __restrict__ x,
                                                       const float* __restrict__ gw,
                                                       float* __restrict__ scores) {
    __shared__ float red[256];
    int t    = blockIdx.x;
    int tid  = threadIdx.x;
    int e    = tid >> 3;
    int part = tid & 7;
    const float* xr = x  + (size_t)t * H_DIM;
    const float* wr = gw + (size_t)e * H_DIM;
    float s = 0.f;
    int h0 = part * (H_DIM / 8);
    #pragma unroll 4
    for (int h = h0; h < h0 + (H_DIM / 8); ++h) s += xr[h] * wr[h];
    red[tid] = s;
    __syncthreads();
    if (part == 0) {
        float tot = 0.f;
        #pragma unroll
        for (int i = 0; i < 8; ++i) tot += red[tid + i];
        scores[(size_t)t * E_EXP + e] = fast_sigmoid(tot);
    }
}

// ---------------- kernel: top-8 select, renormalize, scale -> dense_w[T,E] ----------------
__global__ __launch_bounds__(256) void k_topk(const float* __restrict__ scores,
                                              float* __restrict__ dw) {
    int t = blockIdx.x * 256 + threadIdx.x;
    if (t >= T_TOK) return;
    float s[E_EXP];
    bool  sel[E_EXP];
    #pragma unroll
    for (int e = 0; e < E_EXP; ++e) { s[e] = scores[(size_t)t * E_EXP + e]; sel[e] = false; }
    float sum = 0.f;
    #pragma unroll
    for (int k = 0; k < TOPK; ++k) {
        int best = 0; float bv = -1e30f;
        #pragma unroll
        for (int e = 0; e < E_EXP; ++e)
            if (!sel[e] && s[e] > bv) { bv = s[e]; best = e; }
        sel[best] = true; sum += bv;
    }
    float inv = ROUTED_SCALING * __builtin_amdgcn_rcpf(sum + 1e-20f);
    #pragma unroll
    for (int e = 0; e < E_EXP; ++e)
        dw[(size_t)t * E_EXP + e] = sel[e] ? s[e] * inv : 0.f;
}

// ---------------- kernel: fused gate+up GEMM + SiLU*u -> bf16 activations ----------------
// All operands bf16, B pre-transposed [N,K]. No LDS, no in-loop barriers.
// Block: 256 threads / 8 waves; tile 128(M) x 64(N); wave tile 32x32 (2x2 WMMA) for G and U.
__global__ __launch_bounds__(256) void k_gate_up(const unsigned short* __restrict__ A,   // [M,K]
                                                 const unsigned short* __restrict__ BgT, // [N,K]
                                                 const unsigned short* __restrict__ BuT, // [N,K]
                                                 unsigned short* __restrict__ Hout,      // [M,N]
                                                 int N, int K) {
    const int lane = threadIdx.x & 31;
    const int wave = threadIdx.x >> 5;
    const int wm   = wave & 3;
    const int wn   = wave >> 2;
    const int bM   = blockIdx.x * 128;
    const int bN   = blockIdx.y * 64;
    const int rA0 = bM + wm * 32, rA1 = rA0 + 16;
    const int nB0 = bN + wn * 32, nB1 = nB0 + 16;

    v8f accG[2][2] = {{{0.f}}}, accU[2][2] = {{{0.f}}};

    for (int k0 = 0; k0 < K; k0 += 32) {
        // prefetch 2 K-steps ahead into all cache levels (WGP scope)
        __builtin_prefetch(A   + (size_t)(rA0 + (lane & 15)) * K + k0 + 64, 0, 3);
        __builtin_prefetch(BgT + (size_t)(nB0 + (lane & 15)) * K + k0 + 64, 0, 3);
        __builtin_prefetch(BuT + (size_t)(nB0 + (lane & 15)) * K + k0 + 64, 0, 3);

        v16bf a0  = load_a_frag(A,   K, rA0, k0, lane);
        v16bf a1  = load_a_frag(A,   K, rA1, k0, lane);
        v16bf bg0 = load_b_frag(BgT, K, nB0, k0, lane);
        v16bf bg1 = load_b_frag(BgT, K, nB1, k0, lane);
        v16bf bu0 = load_b_frag(BuT, K, nB0, k0, lane);
        v16bf bu1 = load_b_frag(BuT, K, nB1, k0, lane);

        accG[0][0] = wmma_bf16(a0, bg0, accG[0][0]);
        accG[0][1] = wmma_bf16(a0, bg1, accG[0][1]);
        accG[1][0] = wmma_bf16(a1, bg0, accG[1][0]);
        accG[1][1] = wmma_bf16(a1, bg1, accG[1][1]);
        accU[0][0] = wmma_bf16(a0, bu0, accU[0][0]);
        accU[0][1] = wmma_bf16(a0, bu1, accU[0][1]);
        accU[1][0] = wmma_bf16(a1, bu0, accU[1][0]);
        accU[1][1] = wmma_bf16(a1, bu1, accU[1][1]);
    }

    // epilogue: h = silu(g) * u   (C layout: lane<16 -> N=lane, M=r; lane>=16 -> M=r+8)
    int rbase = bM + wm * 32 + ((lane >> 4) << 3);
    int cbase = bN + wn * 32 + (lane & 15);
    #pragma unroll
    for (int mt = 0; mt < 2; ++mt)
        #pragma unroll
        for (int r = 0; r < 8; ++r) {
            int row = rbase + mt * 16 + r;
            #pragma unroll
            for (int nt = 0; nt < 2; ++nt) {
                float g = accG[mt][nt][r];
                float u = accU[mt][nt][r];
                float h = g * fast_sigmoid(g) * u;
                Hout[(size_t)row * N + cbase + nt * 16] = f32_to_bf16(h);
            }
        }
}

// ---------------- kernel: down-proj GEMM, scaled accumulate into out ----------------
// Block tile 128(M) x 128(N); 8 waves as 4(M) x 2(N); wave tile 32x64 (2x4 WMMA tiles).
// 8 WMMA per 12 b128 loads per wave-step (same intensity as gate_up).
__global__ __launch_bounds__(256) void k_down(const unsigned short* __restrict__ A,   // [M,K]
                                              const unsigned short* __restrict__ BdT, // [N,K]
                                              float* __restrict__ Out,                // [M,N]
                                              const float* __restrict__ Dw,           // [M,E] or null
                                              int expert, int accumulate,
                                              int N, int K) {
    const int lane = threadIdx.x & 31;
    const int wave = threadIdx.x >> 5;
    const int wm   = wave & 3;
    const int wn   = wave >> 2;
    const int bM   = blockIdx.x * 128;
    const int bN   = blockIdx.y * 128;
    const int rA0 = bM + wm * 32, rA1 = rA0 + 16;
    const int nB  = bN + wn * 64;

    v8f acc[2][4] = {{{0.f}}};

    for (int k0 = 0; k0 < K; k0 += 32) {
        __builtin_prefetch(A   + (size_t)(rA0 + (lane & 15)) * K + k0 + 64, 0, 3);
        __builtin_prefetch(BdT + (size_t)(nB  + (lane & 15)) * K + k0 + 64, 0, 3);
        __builtin_prefetch(BdT + (size_t)(nB + 32 + (lane & 15)) * K + k0 + 64, 0, 3);

        v16bf a0 = load_a_frag(A, K, rA0, k0, lane);
        v16bf a1 = load_a_frag(A, K, rA1, k0, lane);
        v16bf b0 = load_b_frag(BdT, K, nB,      k0, lane);
        v16bf b1 = load_b_frag(BdT, K, nB + 16, k0, lane);
        v16bf b2 = load_b_frag(BdT, K, nB + 32, k0, lane);
        v16bf b3 = load_b_frag(BdT, K, nB + 48, k0, lane);

        acc[0][0] = wmma_bf16(a0, b0, acc[0][0]);
        acc[0][1] = wmma_bf16(a0, b1, acc[0][1]);
        acc[0][2] = wmma_bf16(a0, b2, acc[0][2]);
        acc[0][3] = wmma_bf16(a0, b3, acc[0][3]);
        acc[1][0] = wmma_bf16(a1, b0, acc[1][0]);
        acc[1][1] = wmma_bf16(a1, b1, acc[1][1]);
        acc[1][2] = wmma_bf16(a1, b2, acc[1][2]);
        acc[1][3] = wmma_bf16(a1, b3, acc[1][3]);
    }

    int rbase = bM + wm * 32 + ((lane >> 4) << 3);
    int cbase = nB + (lane & 15);
    #pragma unroll
    for (int mt = 0; mt < 2; ++mt)
        #pragma unroll
        for (int r = 0; r < 8; ++r) {
            int row = rbase + mt * 16 + r;
            float s = Dw ? Dw[(size_t)row * E_EXP + expert] : 1.0f;
            #pragma unroll
            for (int nt = 0; nt < 4; ++nt) {
                float v = acc[mt][nt][r] * s;
                size_t o = (size_t)row * N + cbase + nt * 16;
                if (accumulate) Out[o] += v; else Out[o] = v;
            }
        }
}

// ---------------- host-side orchestration ----------------
extern "C" void kernel_launch(void* const* d_in, const int* in_sizes, int n_in,
                              void* d_out, int out_size, void* d_ws, size_t ws_size,
                              hipStream_t stream) {
    (void)in_sizes; (void)n_in; (void)out_size; (void)ws_size;

    const float* x       = (const float*)d_in[0]; // [T, H]
    const float* gate_w  = (const float*)d_in[1]; // [E, H]
    const float* w_gate  = (const float*)d_in[2]; // [E, H, I]
    const float* w_up    = (const float*)d_in[3]; // [E, H, I]
    const float* w_down  = (const float*)d_in[4]; // [E, I, H]
    const float* ws_gate = (const float*)d_in[5]; // [H, IS]
    const float* ws_up   = (const float*)d_in[6]; // [H, IS]
    const float* ws_down = (const float*)d_in[7]; // [IS, H]
    float* out = (float*)d_out;                   // [T, H]

    // workspace layout
    char* ws = (char*)d_ws;
    unsigned short* x_bf  = (unsigned short*)ws;  ws += (size_t)T_TOK * H_DIM * 2;   // 8 MB
    float* scores         = (float*)ws;           ws += (size_t)T_TOK * E_EXP * 4;
    float* dense_w        = (float*)ws;           ws += (size_t)T_TOK * E_EXP * 4;
    unsigned short* h_buf = (unsigned short*)ws;  ws += (size_t)T_TOK * I_DIM * 2;   // 4 MB
    unsigned short* s_act = (unsigned short*)ws;  ws += (size_t)T_TOK * IS_DIM * 2;  // 8 MB
    // transposed bf16 weight staging (sized for the largest user: H*IS)
    unsigned short* wt_g  = (unsigned short*)ws;  ws += (size_t)H_DIM * IS_DIM * 2;  // 8 MB
    unsigned short* wt_u  = (unsigned short*)ws;  ws += (size_t)H_DIM * IS_DIM * 2;  // 8 MB
    unsigned short* wt_d  = (unsigned short*)ws;  ws += (size_t)IS_DIM * H_DIM * 2;  // 8 MB

    dim3 blk(256);

    // 1) x -> bf16
    {
        int n4 = (T_TOK * H_DIM) / 4;
        k_convert_bf16<<<(n4 + 255) / 256, 256, 0, stream>>>((const float4*)x,
                                                             (uint2*)x_bf, n4);
    }
    // 2) router
    k_router_scores<<<T_TOK, 256, 0, stream>>>(x, gate_w, scores);
    k_topk<<<(T_TOK + 255) / 256, 256, 0, stream>>>(scores, dense_w);

    // 3) shared expert (writes every element of d_out first)
    {
        dim3 gt(IS_DIM / 64, H_DIM / 64);
        k_transpose_bf16<<<gt, blk, 0, stream>>>(ws_gate, wt_g, H_DIM, IS_DIM);
        k_transpose_bf16<<<gt, blk, 0, stream>>>(ws_up,   wt_u, H_DIM, IS_DIM);
        dim3 g1(T_TOK / 128, IS_DIM / 64);
        k_gate_up<<<g1, blk, 0, stream>>>(x_bf, wt_g, wt_u, s_act, IS_DIM, H_DIM);
        dim3 gt2(H_DIM / 64, IS_DIM / 64);
        k_transpose_bf16<<<gt2, blk, 0, stream>>>(ws_down, wt_d, IS_DIM, H_DIM);
        dim3 g2(T_TOK / 128, H_DIM / 128);
        k_down<<<g2, blk, 0, stream>>>(s_act, wt_d, out, nullptr, 0, /*accum=*/0,
                                       H_DIM, IS_DIM);
    }
    // 4) routed experts: out += dense_w[:,e] * (silu(x@Wg)*(x@Wu) @ Wd)
    for (int e = 0; e < E_EXP; ++e) {
        const float* Wg = w_gate + (size_t)e * H_DIM * I_DIM;
        const float* Wu = w_up   + (size_t)e * H_DIM * I_DIM;
        const float* Wd = w_down + (size_t)e * I_DIM * H_DIM;
        dim3 gt(I_DIM / 64, H_DIM / 64);
        k_transpose_bf16<<<gt, blk, 0, stream>>>(Wg, wt_g, H_DIM, I_DIM);
        k_transpose_bf16<<<gt, blk, 0, stream>>>(Wu, wt_u, H_DIM, I_DIM);
        dim3 g1(T_TOK / 128, I_DIM / 64);
        k_gate_up<<<g1, blk, 0, stream>>>(x_bf, wt_g, wt_u, h_buf, I_DIM, H_DIM);
        dim3 gt2(H_DIM / 64, I_DIM / 64);
        k_transpose_bf16<<<gt2, blk, 0, stream>>>(Wd, wt_d, I_DIM, H_DIM);
        dim3 g2(T_TOK / 128, H_DIM / 128);
        k_down<<<g2, blk, 0, stream>>>(h_buf, wt_d, out, dense_w, e, /*accum=*/1,
                                       H_DIM, I_DIM);
    }
}
